// GATMLPNet_6957847019826
// MI455X (gfx1250) — matrix operand order repaired and checked by reading, not compiled, not run on hardware
//
#include <hip/hip_runtime.h>
#include <hip/hip_bf16.h>

typedef __attribute__((ext_vector_type(16))) _Float16 v16h;
typedef __attribute__((ext_vector_type(8)))  _Float16 v8h;
typedef __attribute__((ext_vector_type(8)))  float    v8f;

#define NNODE 50000
#define NEDGE 1600000
#define NETOT (NNODE + NEDGE)
#define NGRPH 1024
#define NEG_SLOPE 0.2f
// key of -inf under the monotonic float->uint mapping below
#define NEG_INF_KEY 0x007FFFFFu

// ---------------- helpers ----------------
__device__ __forceinline__ unsigned f32key(float f) {
  unsigned u = __float_as_uint(f);
  return (u & 0x80000000u) ? ~u : (u | 0x80000000u);
}
__device__ __forceinline__ float keyf32(unsigned k) {
  return (k & 0x80000000u) ? __uint_as_float(k & 0x7fffffffu)
                           : __uint_as_float(~k);
}

// ---------------- fills ----------------
__global__ void fill_f32(float* __restrict__ p, long long n, float v) {
  long long i = (long long)blockIdx.x * blockDim.x + threadIdx.x;
  if (i < n) p[i] = v;
}
__global__ void fill_u32(unsigned* __restrict__ p, long long n, unsigned v) {
  long long i = (long long)blockIdx.x * blockDim.x + threadIdx.x;
  if (i < n) p[i] = v;
}

// ---------------- GAT layer 1 node transform + attention dots ----------------
// x:[N,1], W1:[1,128], attS/attD:[2,64] flat(128). One block (128 thr) per node.
__global__ void gat1_node(const float* __restrict__ x, const float* __restrict__ W1,
                          const float* __restrict__ attS, const float* __restrict__ attD,
                          float* __restrict__ h1lin, float* __restrict__ aS,
                          float* __restrict__ aD) {
  __shared__ float ss[128], sd[128];
  int n = blockIdx.x;
  int c = threadIdx.x;
  float hv = x[n] * W1[c];
  h1lin[(size_t)n * 128 + c] = hv;
  ss[c] = hv * attS[c];
  sd[c] = hv * attD[c];
  __syncthreads();
  for (int off = 32; off >= 1; off >>= 1) {
    if ((c & 63) < off) { ss[c] += ss[c + off]; sd[c] += sd[c + off]; }
    __syncthreads();
  }
  if ((c & 63) == 0) {
    int hh = c >> 6;
    aS[n * 2 + hh] = ss[c];
    aD[n * 2 + hh] = sd[c];
  }
}

// ---------------- edge passes (generic over H = 1<<hshift heads) ----------------
__global__ void edge_max(const int* __restrict__ ei0, const int* __restrict__ ei1,
                         const float* __restrict__ aS, const float* __restrict__ aD,
                         unsigned* __restrict__ key, int nE, int nEtot,
                         int hshift, int hmask) {
  int idx = blockIdx.x * blockDim.x + threadIdx.x;
  int total = nEtot << hshift;
  if (idx >= total) return;
  int e = idx >> hshift, h = idx & hmask, H = hmask + 1;
  int s = (e < nE) ? ei0[e] : (e - nE);
  int d = (e < nE) ? ei1[e] : (e - nE);
  float v = aS[s * H + h] + aD[d * H + h];
  v = (v >= 0.f) ? v : NEG_SLOPE * v;
  atomicMax(&key[d * H + h], f32key(v));
}

__global__ void edge_exp(const int* __restrict__ ei0, const int* __restrict__ ei1,
                         const float* __restrict__ aS, const float* __restrict__ aD,
                         const unsigned* __restrict__ key, float* __restrict__ eexp,
                         float* __restrict__ esum, int nE, int nEtot,
                         int hshift, int hmask) {
  int idx = blockIdx.x * blockDim.x + threadIdx.x;
  int total = nEtot << hshift;
  if (idx >= total) return;
  int e = idx >> hshift, h = idx & hmask, H = hmask + 1;
  int s = (e < nE) ? ei0[e] : (e - nE);
  int d = (e < nE) ? ei1[e] : (e - nE);
  float v = aS[s * H + h] + aD[d * H + h];
  v = (v >= 0.f) ? v : NEG_SLOPE * v;
  float m = keyf32(key[d * H + h]);
  float ex = expf(v - m);
  eexp[idx] = ex;
  atomicAdd(&esum[d * H + h], ex);
}

// thread per (edge, feature); F = 1<<fshift total feature channels, C=64 per head
__global__ void edge_scatter(const int* __restrict__ ei0, const int* __restrict__ ei1,
                             const float* __restrict__ eexp, const float* __restrict__ esum,
                             const float* __restrict__ feat, float* __restrict__ acc,
                             int nE, int nEtot, int fshift, int H) {
  long long idx = (long long)blockIdx.x * blockDim.x + threadIdx.x;
  long long total = (long long)nEtot << fshift;
  if (idx >= total) return;
  int e = (int)(idx >> fshift);
  int F = 1 << fshift;
  int f = (int)idx & (F - 1);
  int h = f >> 6;
  int s = (e < nE) ? ei0[e] : (e - nE);
  int d = (e < nE) ? ei1[e] : (e - nE);
  float coeff = eexp[e * H + h] / esum[d * H + h];
  atomicAdd(&acc[(size_t)d * F + f], feat[(size_t)s * F + f] * coeff);
}

// ---------------- epilogues / conversions ----------------
__global__ void finalize_relu_h(const float* __restrict__ acc, const float* __restrict__ bias,
                                _Float16* __restrict__ outH, long long n, int F) {
  long long i = (long long)blockIdx.x * blockDim.x + threadIdx.x;
  if (i >= n) return;
  float v = acc[i] + bias[(int)(i % F)];
  v = fmaxf(v, 0.f);
  outH[i] = (_Float16)v;
}
__global__ void convert_f16(const float* __restrict__ in, _Float16* __restrict__ out,
                            long long n) {
  long long i = (long long)blockIdx.x * blockDim.x + threadIdx.x;
  if (i < n) out[i] = (_Float16)in[i];
}
// W:[K,N] (row-major) -> Wt:[N,K] f16
__global__ void transpose_f16(const float* __restrict__ W, _Float16* __restrict__ Wt,
                              int K, int N) {
  int i = blockIdx.x * blockDim.x + threadIdx.x;
  if (i >= K * N) return;
  int n = i / K, k = i % K;
  Wt[(size_t)n * K + k] = (_Float16)W[(size_t)k * N + n];
}

// ---------------- WMMA GEMM: C[M,64] = act(A[M,K] @ Bt[64,K]^T + bias) ----------------
// A row-major f16, Bt = B transposed (N-major) f16. M multiple of 16; K multiple of 32.
// One wave (32 threads) computes a full 16x64 strip: A registers reused across 4
// accumulators -> 4 back-to-back v_wmma_f32_16x16x32_f16 per K-step (16 for K=128).
__global__ void wmma_gemm_f16_n64(const _Float16* __restrict__ A,
                                  const _Float16* __restrict__ Bt,
                                  const float* __restrict__ bias, float* __restrict__ outF,
                                  _Float16* __restrict__ outH, int M, int K, int relu) {
  const int N = 64;
  int lane = threadIdx.x;
  int mtile = blockIdx.x;
  int half16 = lane >> 4;        // 0: lanes 0-15, 1: lanes 16-31
  int l15 = lane & 15;
  int rowA = mtile * 16 + l15; if (rowA >= M) rowA = M - 1;
  const _Float16* arow = A + (size_t)rowA * K;
  const int kA = half16 * 16;    // A: lanes 0-15 -> K 0..15, lanes 16-31 -> K 16..31
  const int kB = half16 * 16;    // B: lanes 0-15 -> K 0..15, lanes 16-31 -> K 16..31
  // per-lane base pointers for the 4 N-tiles of Bt
  const _Float16* brow0 = Bt + (size_t)(0 * 16 + l15) * K + kB;
  const _Float16* brow1 = Bt + (size_t)(1 * 16 + l15) * K + kB;
  const _Float16* brow2 = Bt + (size_t)(2 * 16 + l15) * K + kB;
  const _Float16* brow3 = Bt + (size_t)(3 * 16 + l15) * K + kB;
  union { v16h v; v8h h[2]; } a;
  v8f acc0 = {}, acc1 = {}, acc2 = {}, acc3 = {};
  for (int k0 = 0; k0 < K; k0 += 32) {
    a.h[0] = *(const v8h*)(arow + k0 + kA);
    a.h[1] = *(const v8h*)(arow + k0 + kA + 8);
    v16h b0 = *(const v16h*)(brow0 + k0);
    v16h b1 = *(const v16h*)(brow1 + k0);
    v16h b2 = *(const v16h*)(brow2 + k0);
    v16h b3 = *(const v16h*)(brow3 + k0);
    acc0 = __builtin_amdgcn_wmma_f32_16x16x32_f16(false, a.v, false, b0, (short)0, acc0, false, false);
    acc1 = __builtin_amdgcn_wmma_f32_16x16x32_f16(false, a.v, false, b1, (short)0, acc1, false, false);
    acc2 = __builtin_amdgcn_wmma_f32_16x16x32_f16(false, a.v, false, b2, (short)0, acc2, false, false);
    acc3 = __builtin_amdgcn_wmma_f32_16x16x32_f16(false, a.v, false, b3, (short)0, acc3, false, false);
  }
  v8f accs[4] = {acc0, acc1, acc2, acc3};
#pragma unroll
  for (int j = 0; j < 4; ++j) {
    int col = j * 16 + l15;
    float bv = bias ? bias[col] : 0.f;
#pragma unroll
    for (int r = 0; r < 8; ++r) {
      int row = mtile * 16 + r + half16 * 8;  // C/D: VGPR r -> M=r (+8 for hi lanes)
      if (row >= M) continue;
      float v = accs[j][r] + bv;
      if (relu) v = fmaxf(v, 0.f);
      if (outF) outF[(size_t)row * N + col] = v;
      if (outH) outH[(size_t)row * N + col] = (_Float16)v;
    }
  }
}

// ---------------- layer-2 attention dots (H=1, C=64) ----------------
__global__ void alpha_node(const float* __restrict__ t2, const float* __restrict__ attS,
                           const float* __restrict__ attD, float* __restrict__ aS,
                           float* __restrict__ aD, int n) {
  int i = blockIdx.x * blockDim.x + threadIdx.x;
  if (i >= n) return;
  const float* row = t2 + (size_t)i * 64;
  float s = 0.f, d = 0.f;
  for (int c = 0; c < 64; ++c) { s += row[c] * attS[c]; d += row[c] * attD[c]; }
  aS[i] = s; aD[i] = d;
}

// ---------------- mean pool + head ----------------
__global__ void pool_accum(const float* __restrict__ acc2, const float* __restrict__ b2,
                           const int* __restrict__ batch, float* __restrict__ gsum) {
  long long i = (long long)blockIdx.x * blockDim.x + threadIdx.x;
  if (i >= (long long)NNODE * 64) return;
  int n = (int)(i >> 6), c = (int)i & 63;
  atomicAdd(&gsum[(size_t)batch[n] * 64 + c], acc2[i] + b2[c]);
}
__global__ void count_nodes(const int* __restrict__ batch, float* __restrict__ gcnt) {
  int i = blockIdx.x * blockDim.x + threadIdx.x;
  if (i >= NNODE) return;
  atomicAdd(&gcnt[batch[i]], 1.f);
}
__global__ void final_fc(const float* __restrict__ gsum, const float* __restrict__ gcnt,
                         const float* __restrict__ mlpo, const float* __restrict__ FCw,
                         const float* __restrict__ FCb, float* __restrict__ out) {
  int idx = blockIdx.x * blockDim.x + threadIdx.x;
  if (idx >= NGRPH * 2) return;
  int g = idx >> 1, o = idx & 1;
  float inv = 1.f / fmaxf(gcnt[g], 1.f);
  float acc = FCb[o];
  for (int c = 0; c < 64; ++c) acc += (gsum[g * 64 + c] * inv) * FCw[c * 2 + o];
  for (int c = 0; c < 64; ++c) acc += mlpo[g * 64 + c] * FCw[(64 + c) * 2 + o];
  out[idx] = acc;
}

// ---------------- host launch ----------------
static inline int gdiv(long long n, int b) { return (int)((n + b - 1) / b); }

extern "C" void kernel_launch(void* const* d_in, const int* in_sizes, int n_in,
                              void* d_out, int out_size, void* d_ws, size_t ws_size,
                              hipStream_t stream) {
  const float* x    = (const float*)d_in[0];
  const int*   ei0  = (const int*)d_in[1];
  const int*   ei1  = ei0 + NEDGE;
  const int*   batch= (const int*)d_in[2];
  const float* nd   = (const float*)d_in[3];
  const float* W1   = (const float*)d_in[4];
  const float* as1  = (const float*)d_in[5];
  const float* ad1  = (const float*)d_in[6];
  const float* b1   = (const float*)d_in[7];
  const float* W2   = (const float*)d_in[8];
  const float* as2  = (const float*)d_in[9];
  const float* ad2  = (const float*)d_in[10];
  const float* b2   = (const float*)d_in[11];
  const float* M1w  = (const float*)d_in[12];
  const float* M1b  = (const float*)d_in[13];
  const float* M2w  = (const float*)d_in[14];
  const float* M2b  = (const float*)d_in[15];
  const float* M3w  = (const float*)d_in[16];
  const float* M3b  = (const float*)d_in[17];
  const float* FCw  = (const float*)d_in[18];
  const float* FCb  = (const float*)d_in[19];
  float* out = (float*)d_out;

  // ---- workspace bump allocator (256B aligned) ----
  char* W = (char*)d_ws;
  size_t off = 0;
  auto take = [&](size_t bytes) -> char* {
    char* p = W + off;
    off = (off + bytes + 255) & ~(size_t)255;
    return p;
  };
  float*    h1lin = (float*)take((size_t)NNODE * 128 * 4);  // reused as t2 in layer 2
  float*    acc1  = (float*)take((size_t)NNODE * 128 * 4);  // reused as acc2 in layer 2
  float*    aS    = (float*)take((size_t)NNODE * 2 * 4);
  float*    aD    = (float*)take((size_t)NNODE * 2 * 4);
  unsigned* key   = (unsigned*)take((size_t)NNODE * 2 * 4);
  float*    esum  = (float*)take((size_t)NNODE * 2 * 4);
  float*    eexp  = (float*)take((size_t)NETOT * 2 * 4);
  _Float16* h1h   = (_Float16*)take((size_t)NNODE * 128 * 2);
  _Float16* w2t   = (_Float16*)take((size_t)64 * 128 * 2);
  float*    gsum  = (float*)take((size_t)NGRPH * 64 * 4);
  float*    gcnt  = (float*)take((size_t)NGRPH * 4);
  _Float16* ndh   = (_Float16*)take((size_t)NGRPH * 128 * 2);
  _Float16* m1t   = (_Float16*)take((size_t)64 * 128 * 2);
  _Float16* m2t   = (_Float16*)take((size_t)64 * 64 * 2);
  _Float16* m3t   = (_Float16*)take((size_t)64 * 64 * 2);
  _Float16* mb1   = (_Float16*)take((size_t)NGRPH * 64 * 2);
  _Float16* mb2   = (_Float16*)take((size_t)NGRPH * 64 * 2);
  float*    mlpo  = (float*)take((size_t)NGRPH * 64 * 4);
  float*    t2    = h1lin;   // layer-2 reuse (phase-ordered on stream)
  float*    acc2  = acc1;
  (void)ws_size; (void)in_sizes; (void)n_in; (void)out_size;

  // ======== GAT layer 1 ========
  gat1_node<<<NNODE, 128, 0, stream>>>(x, W1, as1, ad1, h1lin, aS, aD);
  fill_u32<<<gdiv((long long)NNODE * 2, 256), 256, 0, stream>>>(key, (long long)NNODE * 2, NEG_INF_KEY);
  fill_f32<<<gdiv((long long)NNODE * 2, 256), 256, 0, stream>>>(esum, (long long)NNODE * 2, 0.f);
  fill_f32<<<gdiv((long long)NNODE * 128, 256), 256, 0, stream>>>(acc1, (long long)NNODE * 128, 0.f);
  edge_max<<<gdiv((long long)NETOT * 2, 256), 256, 0, stream>>>(ei0, ei1, aS, aD, key, NEDGE, NETOT, 1, 1);
  edge_exp<<<gdiv((long long)NETOT * 2, 256), 256, 0, stream>>>(ei0, ei1, aS, aD, key, eexp, esum, NEDGE, NETOT, 1, 1);
  edge_scatter<<<gdiv((long long)NETOT * 128, 256), 256, 0, stream>>>(ei0, ei1, eexp, esum, h1lin, acc1, NEDGE, NETOT, 7, 2);
  finalize_relu_h<<<gdiv((long long)NNODE * 128, 256), 256, 0, stream>>>(acc1, b1, h1h, (long long)NNODE * 128, 128);

  // ======== GAT layer 2: t2 = h1 @ W2 via WMMA, then attention + scatter ========
  transpose_f16<<<gdiv(128 * 64, 256), 256, 0, stream>>>(W2, w2t, 128, 64);
  wmma_gemm_f16_n64<<<NNODE / 16, 32, 0, stream>>>(h1h, w2t, nullptr, t2, nullptr, NNODE, 128, 0);
  alpha_node<<<gdiv(NNODE, 256), 256, 0, stream>>>(t2, as2, ad2, aS, aD, NNODE);
  fill_u32<<<gdiv(NNODE, 256), 256, 0, stream>>>(key, NNODE, NEG_INF_KEY);
  fill_f32<<<gdiv(NNODE, 256), 256, 0, stream>>>(esum, NNODE, 0.f);
  fill_f32<<<gdiv((long long)NNODE * 64, 256), 256, 0, stream>>>(acc2, (long long)NNODE * 64, 0.f);
  edge_max<<<gdiv(NETOT, 256), 256, 0, stream>>>(ei0, ei1, aS, aD, key, NEDGE, NETOT, 0, 0);
  edge_exp<<<gdiv(NETOT, 256), 256, 0, stream>>>(ei0, ei1, aS, aD, key, eexp, esum, NEDGE, NETOT, 0, 0);
  edge_scatter<<<gdiv((long long)NETOT * 64, 256), 256, 0, stream>>>(ei0, ei1, eexp, esum, t2, acc2, NEDGE, NETOT, 6, 1);

  // ======== mean pool ========
  fill_f32<<<gdiv(NGRPH * 64, 256), 256, 0, stream>>>(gsum, NGRPH * 64, 0.f);
  fill_f32<<<gdiv(NGRPH, 256), 256, 0, stream>>>(gcnt, NGRPH, 0.f);
  pool_accum<<<gdiv((long long)NNODE * 64, 256), 256, 0, stream>>>(acc2, b2, batch, gsum);
  count_nodes<<<gdiv(NNODE, 256), 256, 0, stream>>>(batch, gcnt);

  // ======== MLP branch (all WMMA) ========
  convert_f16<<<gdiv(NGRPH * 128, 256), 256, 0, stream>>>(nd, ndh, NGRPH * 128);
  transpose_f16<<<gdiv(128 * 64, 256), 256, 0, stream>>>(M1w, m1t, 128, 64);
  transpose_f16<<<gdiv(64 * 64, 256), 256, 0, stream>>>(M2w, m2t, 64, 64);
  transpose_f16<<<gdiv(64 * 64, 256), 256, 0, stream>>>(M3w, m3t, 64, 64);
  wmma_gemm_f16_n64<<<NGRPH / 16, 32, 0, stream>>>(ndh, m1t, M1b, nullptr, mb1, NGRPH, 128, 1);
  wmma_gemm_f16_n64<<<NGRPH / 16, 32, 0, stream>>>(mb1, m2t, M2b, nullptr, mb2, NGRPH, 64, 1);
  wmma_gemm_f16_n64<<<NGRPH / 16, 32, 0, stream>>>(mb2, m3t, M3b, mlpo, nullptr, NGRPH, 64, 0);

  // ======== final head ========
  final_fc<<<gdiv(NGRPH * 2, 256), 256, 0, stream>>>(gsum, gcnt, mlpo, FCw, FCb, out);
}
